// Encoder_44109314130176
// MI455X (gfx1250) — matrix operand order repaired and checked by reading, not compiled, not run on hardware
//
#include <hip/hip_runtime.h>
#include <hip/hip_bf16.h>
#include <stdint.h>

// Problem dims
#define B_ 32
#define S_ 512
#define H_ 1024
#define G_ 4096   // 4*H

typedef __bf16 bf16;
typedef __attribute__((ext_vector_type(16))) __bf16 v16bf;
typedef __attribute__((ext_vector_type(8)))  __bf16 v8bf;
typedef __attribute__((ext_vector_type(8)))  float  v8f;

// ---------------------------------------------------------------------------
// Async copy of a contiguous strip from global memory into LDS.
// CDNA5 GLOBAL_LOAD_ASYNC_TO_LDS_B128 (ASYNCcnt-tracked, no VGPR round trip),
// then s_wait_asynccnt + workgroup barrier.
// ---------------------------------------------------------------------------
__device__ __forceinline__ void stage_strip_async(const bf16* __restrict__ g,
                                                  unsigned lds_off, int nbytes) {
  for (int o = (int)threadIdx.x * 16; o < nbytes; o += (int)blockDim.x * 16) {
    const void* ga = (const void*)((const char*)g + o);
    unsigned la = lds_off + (unsigned)o;
    asm volatile("global_load_async_to_lds_b128 %0, %1, off"
                 :: "v"(la), "v"(ga) : "memory");
  }
  asm volatile("s_wait_asynccnt 0x0" ::: "memory");
  __syncthreads();
}

// ---------------------------------------------------------------------------
// WMMA fragment loaders (wave32, 16x16x32 bf16; contiguous k = 16*(lane/16) + s in both A and B)
// A: 16x32 (MxK) slice from an LDS strip laid out [rows][H_] row-major.
//    lane l holds row rbase + l%16; lanes0-15 K=[0..15],
//    lanes16-31 K=[16..31].
// ---------------------------------------------------------------------------
__device__ __forceinline__ v16bf lds_a_frag(const bf16* As, int rbase, int kb) {
  const int l  = threadIdx.x & 31;
  const bf16* p = As + (rbase + (l & 15)) * H_ + kb + ((l >> 4) * 16);
  v8bf lo = *(const v8bf*)p;
  v8bf hi = *(const v8bf*)(p + 8);
  v16bf r;
#pragma unroll
  for (int i = 0; i < 8; ++i) { r[i] = lo[i]; r[8 + i] = hi[i]; }
  return r;
}

// B: 32x16 (KxN). Source BT is the transposed weight, row-major [N, K], so a
// lane's fragment (one N column, 16 consecutive K) is one contiguous 32B load.
__device__ __forceinline__ v16bf load_b_frag(const bf16* __restrict__ BT, int ldb,
                                             int nb, int kb) {
  const int l  = threadIdx.x & 31;
  const int n  = nb + (l & 15);
  const int ko = (l >> 4) * 16;
  return *(const v16bf*)(BT + (size_t)n * ldb + kb + ko);
}

__device__ __forceinline__ float sigmoidf_(float x) {
  return 1.0f / (1.0f + __expf(-x));
}

// ---------------------------------------------------------------------------
// Init: zero recurrent state + barrier words (ws is poisoned by harness)
// ---------------------------------------------------------------------------
__global__ void init_state_kernel(float* hbuf, float* cbuf, float* opbuf,
                                  bf16* hbf, unsigned* bar) {
  const int n = 2 * B_ * H_;
  const int stride = gridDim.x * blockDim.x;
  for (int i = blockIdx.x * blockDim.x + threadIdx.x; i < n; i += stride) {
    hbuf[i] = 0.0f; cbuf[i] = 0.0f; opbuf[i] = 0.0f; hbf[i] = (bf16)0.0f;
  }
  if (blockIdx.x == 0 && threadIdx.x == 0) { bar[0] = 0u; bar[1] = 0u; }
}

// ---------------------------------------------------------------------------
// fp32 -> bf16 elementwise convert (for x)
// ---------------------------------------------------------------------------
__global__ void convert_bf16_kernel(const float* __restrict__ in,
                                    bf16* __restrict__ out, size_t n) {
  size_t i = (size_t)blockIdx.x * blockDim.x + threadIdx.x;
  const size_t stride = (size_t)gridDim.x * blockDim.x;
  for (; i < n; i += stride) out[i] = (bf16)in[i];
}

// ---------------------------------------------------------------------------
// Tiled transpose + convert: in fp32 [R, C] row-major -> out bf16 [C, R]
// ---------------------------------------------------------------------------
__global__ __launch_bounds__(256) void transpose_bf16_kernel(
    const float* __restrict__ in, bf16* __restrict__ out, int R, int C) {
  __shared__ float tile[32][33];
  const int c0 = blockIdx.x * 32;
  const int r0 = blockIdx.y * 32;
  const int tx = threadIdx.x;  // 0..31
  const int ty = threadIdx.y;  // 0..7
#pragma unroll
  for (int i = 0; i < 32; i += 8) {
    tile[ty + i][tx] = in[(size_t)(r0 + ty + i) * C + (c0 + tx)];
  }
  __syncthreads();
#pragma unroll
  for (int i = 0; i < 32; i += 8) {
    out[(size_t)(c0 + ty + i) * R + (r0 + tx)] = (bf16)tile[tx][ty + i];
  }
}

// ---------------------------------------------------------------------------
// GEMM1: xz[dir] = x @ W[dir] + b[dir]   (M=B*S=16384, K=H, N=4H)
// Block stages a 64-row A strip (128 KB) into LDS via async loads; each of
// the 8 waves computes a 64x32 macro-tile (4 A frags x 2 B frags -> 8 WMMA
// per k-step). Weight panel L2 traffic: 4 GB total (4x less than 16-row tiles).
// ---------------------------------------------------------------------------
__global__ __launch_bounds__(256) void gemm_xz_kernel(
    const bf16* __restrict__ xb,                       // [B*S, H] bf16
    const bf16* __restrict__ WT_fw, const bf16* __restrict__ WT_bw,  // [4H, H]
    const float* __restrict__ b_fw, const float* __restrict__ b_bw,
    float* __restrict__ xz_fw, float* __restrict__ xz_bw) {
  __shared__ bf16 As[64 * H_];  // 128 KB
  const int dir = blockIdx.y;
  const bf16*  WT   = dir ? WT_bw : WT_fw;
  const float* bias = dir ? b_bw  : b_fw;
  float*       xz   = dir ? xz_bw : xz_fw;

  const int mt  = blockIdx.x >> 4;                     // 0..255 (M tiles of 64)
  const int ng  = blockIdx.x & 15;                     // 0..15 (N groups of 256)
  const int mb  = mt * 64;
  const int w   = threadIdx.x >> 5;
  const int nb0 = ng * 256 + w * 32;                   // per-wave 32 columns
  const int l   = threadIdx.x & 31;

  // Stage the 64-row A strip (contiguous 128 KB) into LDS.
  stage_strip_async(xb + (size_t)mb * H_, (unsigned)(uintptr_t)As,
                    64 * H_ * (int)sizeof(bf16));

  v8f acc[4][2];
#pragma unroll
  for (int u = 0; u < 2; ++u) {
    const float bv = bias[nb0 + 16 * u + (l & 15)];
#pragma unroll
    for (int r = 0; r < 4; ++r)
#pragma unroll
      for (int j = 0; j < 8; ++j) acc[r][u][j] = bv;
  }

  // Software-pipelined k-loop (B fragments double-buffered; A from LDS)
  v16bf b0[2];
#pragma unroll
  for (int u = 0; u < 2; ++u) b0[u] = load_b_frag(WT, H_, nb0 + 16 * u, 0);

#pragma unroll 2
  for (int kb = 32; kb < H_; kb += 32) {
    v16bf b1[2];
#pragma unroll
    for (int u = 0; u < 2; ++u) b1[u] = load_b_frag(WT, H_, nb0 + 16 * u, kb);
    v16bf a[4];
#pragma unroll
    for (int r = 0; r < 4; ++r) a[r] = lds_a_frag(As, r * 16, kb - 32);
#pragma unroll
    for (int r = 0; r < 4; ++r)
#pragma unroll
      for (int u = 0; u < 2; ++u)
        acc[r][u] = __builtin_amdgcn_wmma_f32_16x16x32_bf16(
            false, a[r], false, b0[u], (short)0, acc[r][u], false, false);
#pragma unroll
    for (int u = 0; u < 2; ++u) b0[u] = b1[u];
  }
  {
    v16bf a[4];
#pragma unroll
    for (int r = 0; r < 4; ++r) a[r] = lds_a_frag(As, r * 16, H_ - 32);
#pragma unroll
    for (int r = 0; r < 4; ++r)
#pragma unroll
      for (int u = 0; u < 2; ++u)
        acc[r][u] = __builtin_amdgcn_wmma_f32_16x16x32_bf16(
            false, a[r], false, b0[u], (short)0, acc[r][u], false, false);
  }

  // C/D layout: VGPR j -> row base + j + (lane/16)*8 ; col = nb + lane%16
  const int cb = l & 15;
#pragma unroll
  for (int r = 0; r < 4; ++r) {
    const int rowb = mb + r * 16 + (l >> 4) * 8;
#pragma unroll
    for (int u = 0; u < 2; ++u) {
      const int col = nb0 + 16 * u + cb;
#pragma unroll
      for (int j = 0; j < 8; ++j) {
        xz[(size_t)(rowb + j) * G_ + col] = acc[r][u][j];
      }
    }
  }
}

// ---------------------------------------------------------------------------
// Device-wide sense barrier (persistent kernel). bar[0]=counter, bar[1]=gen.
// ---------------------------------------------------------------------------
__device__ __forceinline__ void grid_barrier(unsigned* bar, unsigned nblocks) {
  __syncthreads();
  if (threadIdx.x == 0) {
    const unsigned g = __hip_atomic_load(&bar[1], __ATOMIC_RELAXED,
                                         __HIP_MEMORY_SCOPE_AGENT);
    const unsigned arrived = __hip_atomic_fetch_add(&bar[0], 1u, __ATOMIC_ACQ_REL,
                                                    __HIP_MEMORY_SCOPE_AGENT);
    if (arrived == nblocks - 1) {
      __hip_atomic_store(&bar[0], 0u, __ATOMIC_RELAXED, __HIP_MEMORY_SCOPE_AGENT);
      __hip_atomic_fetch_add(&bar[1], 1u, __ATOMIC_RELEASE,
                             __HIP_MEMORY_SCOPE_AGENT);
    } else {
      while (__hip_atomic_load(&bar[1], __ATOMIC_ACQUIRE,
                               __HIP_MEMORY_SCOPE_AGENT) == g) {
        __builtin_amdgcn_s_sleep(1);
      }
    }
  }
  __syncthreads();
}

// ---------------------------------------------------------------------------
// Persistent recurrent kernel: 512 steps, 2 directions in lockstep.
// 64 blocks x 8 waves = 512 wave-jobs = 2 dirs x 256 N tiles. Each wave
// computes a 32x16 tile of h@U (both batch halves share one B fragment ->
// minimal 16 MB/step U stream from L2). The block's full 32-row h (64 KB)
// is async-staged into LDS every step.
// ---------------------------------------------------------------------------
#define NBLK 64
#define NTHR 256

__global__ __launch_bounds__(NTHR) void lstm_recurrent_kernel(
    const float* __restrict__ xz_fw, const float* __restrict__ xz_bw, // [B,S,4H]
    const bf16* __restrict__ UT_fw,  const bf16* __restrict__ UT_bw,  // [4H,H]
    const float* __restrict__ pm,                                     // [B,S]
    float* __restrict__ hbuf, float* __restrict__ cbuf,               // [2,B,H]
    float* __restrict__ opbuf,                                        // [2,B,H]
    bf16* __restrict__ hbf,                                           // [2,B,H]
    float* __restrict__ zstep,                                        // [2,B,4H]
    float* __restrict__ dout, unsigned* __restrict__ bar) {
  __shared__ bf16 As[32 * H_];  // 64 KB: full h for this block's direction
  const int tid      = blockIdx.x * blockDim.x + threadIdx.x;
  const int nthreads = gridDim.x * blockDim.x;
  const int l        = threadIdx.x & 31;

  // Job assignment: 512 jobs == 512 waves. All 8 waves of a block share dir.
  const int job = blockIdx.x * (NTHR / 32) + (int)(threadIdx.x >> 5);
  const int d   = job >> 8;                 // direction (same across block)
  const int nb  = (job & 255) * 16;         // N tile
  const bf16*  UT  = d ? UT_bw : UT_fw;
  const float* xzd = d ? xz_bw : xz_fw;
  const unsigned AsOff = (unsigned)(uintptr_t)As;

  for (int s = 0; s < S_; ++s) {
    const int t = d ? (S_ - 1 - s) : s;

    // ---- Stage full h for this dir (32 rows x H = 64 KB) into LDS ----
    stage_strip_async(hbf + d * (B_ * H_), AsOff, 32 * H_ * (int)sizeof(bf16));

    // ---- Phase 1: zstep = xz[:,t,:] + h @ U (32x16 tile per wave) ----
    {
      v8f acc0 = {0.f, 0.f, 0.f, 0.f, 0.f, 0.f, 0.f, 0.f};
      v8f acc1 = acc0;
      v16bf b0 = load_b_frag(UT, H_, nb, 0);
#pragma unroll 2
      for (int kb = 32; kb < H_; kb += 32) {
        v16bf b1 = load_b_frag(UT, H_, nb, kb);
        v16bf a0 = lds_a_frag(As, 0, kb - 32);
        v16bf a1 = lds_a_frag(As, 16, kb - 32);
        acc0 = __builtin_amdgcn_wmma_f32_16x16x32_bf16(
            false, a0, false, b0, (short)0, acc0, false, false);
        acc1 = __builtin_amdgcn_wmma_f32_16x16x32_bf16(
            false, a1, false, b0, (short)0, acc1, false, false);
        b0 = b1;
      }
      {
        v16bf a0 = lds_a_frag(As, 0, H_ - 32);
        v16bf a1 = lds_a_frag(As, 16, H_ - 32);
        acc0 = __builtin_amdgcn_wmma_f32_16x16x32_bf16(
            false, a0, false, b0, (short)0, acc0, false, false);
        acc1 = __builtin_amdgcn_wmma_f32_16x16x32_bf16(
            false, a1, false, b0, (short)0, acc1, false, false);
      }

      const int col = nb + (l & 15);
      const int rb  = (l >> 4) * 8;
      float* zs = zstep + d * (B_ * G_);
#pragma unroll
      for (int j = 0; j < 8; ++j) {
        const int bb = rb + j;          // batch rows 0..15
        zs[bb * G_ + col] = acc0[j] + xzd[((size_t)bb * S_ + t) * G_ + col];
      }
#pragma unroll
      for (int j = 0; j < 8; ++j) {
        const int bb = 16 + rb + j;     // batch rows 16..31
        zs[bb * G_ + col] = acc1[j] + xzd[((size_t)bb * S_ + t) * G_ + col];
      }
    }
    grid_barrier(bar, gridDim.x);

    // ---- Phase 2: gates + masked state update (2*B*H = 65536 elems) ----
    for (int i = tid; i < 2 * B_ * H_; i += nthreads) {
      const int dd = i >> 15;
      const int r  = i & 32767;
      const int bb = r >> 10;
      const int hh = r & 1023;
      const int tt = dd ? (S_ - 1 - s) : s;
      const float m = 1.0f - pm[bb * S_ + tt];
      const float* z = zstep + dd * (B_ * G_) + bb * G_;
      const float ig = sigmoidf_(z[hh]);
      const float fg = sigmoidf_(z[H_ + hh]);
      const float gg = tanhf(z[2 * H_ + hh]);
      const float og = sigmoidf_(z[3 * H_ + hh]);
      const float cp = cbuf[i], hp = hbuf[i], op = opbuf[i];
      const float cn = fg * cp + ig * gg;
      const float hn = og * tanhf(cn);
      const float h2  = m * hn + (1.0f - m) * hp;
      const float c2  = m * cn + (1.0f - m) * cp;
      const float out = m * hn + (1.0f - m) * op;
      hbuf[i] = h2; cbuf[i] = c2; opbuf[i] = out;
      hbf[i] = (bf16)h2;
      dout[((size_t)bb * S_ + tt) * (2 * H_) + dd * H_ + hh] = out;
    }
    grid_barrier(bar, gridDim.x);
  }

  // ---- Final states: h_fw, c_fw, h_bw, c_bw after encoder_outputs ----
  const size_t ENC = (size_t)B_ * S_ * 2 * H_;
  for (int i = tid; i < 2 * B_ * H_; i += nthreads) {
    const int dd = i >> 15;
    const int r  = i & 32767;
    dout[ENC + (size_t)dd * 65536 + r]         = hbuf[i];  // h_fw / h_bw
    dout[ENC + (size_t)dd * 65536 + 32768 + r] = cbuf[i];  // c_fw / c_bw
  }
}

// ---------------------------------------------------------------------------
// Host side
// ---------------------------------------------------------------------------
extern "C" void kernel_launch(void* const* d_in, const int* in_sizes, int n_in,
                              void* d_out, int out_size, void* d_ws, size_t ws_size,
                              hipStream_t stream) {
  (void)in_sizes; (void)n_in; (void)out_size; (void)ws_size;
  const float* x    = (const float*)d_in[0];  // [B,S,H]
  const float* pm   = (const float*)d_in[1];  // [B,S]
  const float* W_fw = (const float*)d_in[2];  // [H,4H]
  const float* U_fw = (const float*)d_in[3];  // [H,4H]
  const float* b_fw = (const float*)d_in[4];  // [4H]
  const float* W_bw = (const float*)d_in[5];
  const float* U_bw = (const float*)d_in[6];
  const float* b_bw = (const float*)d_in[7];
  float* dout = (float*)d_out;

  // Workspace carve-up (~578 MB total)
  char* ws = (char*)d_ws;
  size_t off = 0;
  auto alloc = [&](size_t bytes) -> void* {
    void* p = ws + off;
    off += (bytes + 255) & ~(size_t)255;
    return p;
  };
  float* xz_fw = (float*)alloc((size_t)B_ * S_ * G_ * sizeof(float));  // 256 MB
  float* xz_bw = (float*)alloc((size_t)B_ * S_ * G_ * sizeof(float));  // 256 MB
  bf16*  xb    = (bf16*) alloc((size_t)B_ * S_ * H_ * sizeof(bf16));   //  32 MB
  bf16*  WT_fw = (bf16*) alloc((size_t)G_ * H_ * sizeof(bf16));        //   8 MB
  bf16*  WT_bw = (bf16*) alloc((size_t)G_ * H_ * sizeof(bf16));
  bf16*  UT_fw = (bf16*) alloc((size_t)G_ * H_ * sizeof(bf16));
  bf16*  UT_bw = (bf16*) alloc((size_t)G_ * H_ * sizeof(bf16));
  float* hbuf  = (float*)alloc((size_t)2 * B_ * H_ * sizeof(float));
  float* cbuf  = (float*)alloc((size_t)2 * B_ * H_ * sizeof(float));
  float* opbuf = (float*)alloc((size_t)2 * B_ * H_ * sizeof(float));
  bf16*  hbf   = (bf16*) alloc((size_t)2 * B_ * H_ * sizeof(bf16));
  float* zstep = (float*)alloc((size_t)2 * B_ * G_ * sizeof(float));   //   1 MB
  unsigned* bar = (unsigned*)alloc(256);

  // 1) zero recurrent state + barrier
  init_state_kernel<<<64, 256, 0, stream>>>(hbuf, cbuf, opbuf, hbf, bar);

  // 2) x -> bf16
  convert_bf16_kernel<<<2048, 256, 0, stream>>>(x, xb, (size_t)B_ * S_ * H_);

  // 3) transpose-convert weights: [H,4H] fp32 -> [4H,H] bf16
  dim3 tb(32, 8);
  dim3 tg(G_ / 32, H_ / 32);
  transpose_bf16_kernel<<<tg, tb, 0, stream>>>(W_fw, WT_fw, H_, G_);
  transpose_bf16_kernel<<<tg, tb, 0, stream>>>(W_bw, WT_bw, H_, G_);
  transpose_bf16_kernel<<<tg, tb, 0, stream>>>(U_fw, UT_fw, H_, G_);
  transpose_bf16_kernel<<<tg, tb, 0, stream>>>(U_bw, UT_bw, H_, G_);

  // 4) xz = x @ W + b: blocks = 256 Mtiles x 16 Ngroups, y = direction
  dim3 gg(4096, 2);
  gemm_xz_kernel<<<gg, 256, 0, stream>>>(xb, WT_fw, WT_bw, b_fw, b_bw,
                                         xz_fw, xz_bw);

  // 5) persistent recurrence (64 blocks * 8 waves = 512 wave-jobs)
  lstm_recurrent_kernel<<<NBLK, NTHR, 0, stream>>>(
      xz_fw, xz_bw, UT_fw, UT_bw, pm, hbuf, cbuf, opbuf, hbf, zstep, dout, bar);
}
